// LoRALinear_3350074491354
// MI455X (gfx1250) — matrix operand order repaired and checked by reading, not compiled, not run on hardware
//
#include <hip/hip_runtime.h>

// ---------------------------------------------------------------------------
// QLoRA forward for MI455X (gfx1250):
//   out = x @ (dequant_nf4(W) + (alpha/rank) * B @ A)^T
// Fold LoRA into dequantized bf16 weight once, convert x to bf16 once, then a
// single compute-bound WMMA bf16 GEMM with double-buffered LDS tiles filled by
// CDNA5 async global->LDS DMA (global_load_async_to_lds_b128 / ASYNCcnt).
// Buffer toggle is a loop-carried dynamic offset so the K loop has ONE body
// and accumulators keep a stable register assignment (no v_mov/v_nop churn).
// ---------------------------------------------------------------------------

typedef __attribute__((ext_vector_type(16))) __bf16 v16bf;
typedef __attribute__((ext_vector_type(8)))  float  v8f;

#define IN_DIM   4096
#define OUT_DIM  4096
#define M_TOTAL  8192          // 4 * 2048
#define RANK     16
#define LORA_SCALE 2.0f        // alpha / rank = 32 / 16

#define BM 128
#define BN 128
#define BK 32
#define KSTEPS (IN_DIM / BK)
#define XBUF (BM * BK)         // elements per X tile buffer
#define WBUF (BN * BK)         // elements per W tile buffer

__constant__ float NF4_TAB[16] = {
    -1.0f, -0.6961928009986877f, -0.5250730514526367f, -0.39491748809814453f,
    -0.28444138169288635f, -0.18477343022823334f, -0.09105003625154495f, 0.0f,
    0.07958029955625534f, 0.16093020141124725f, 0.24611230194568634f,
    0.33791524171829224f, 0.44070982933044434f, 0.5626170039176941f,
    0.7229568362236023f, 1.0f };

// ---------------------------------------------------------------------------
// Kernel 1: dequantize NF4 weight (double-quantized scalers) and fold in the
// rank-16 LoRA update.  One element per thread; W_eff written as bf16.
// ---------------------------------------------------------------------------
__global__ __launch_bounds__(256)
void dequant_fold_kernel(const int*   __restrict__ nf4_idx,
                         const int*   __restrict__ q_scalers,
                         const float* __restrict__ q_factor,
                         const float* __restrict__ scaler_mean,
                         const float* __restrict__ lora_a,    // [RANK, IN_DIM]
                         const float* __restrict__ lora_b,    // [OUT_DIM, RANK]
                         __bf16*      __restrict__ weff)      // [OUT_DIM, IN_DIM]
{
    unsigned idx = blockIdx.x * blockDim.x + threadIdx.x;   // 0 .. 16M-1
    unsigned o   = idx >> 12;                               // / IN_DIM
    unsigned i   = idx & (IN_DIM - 1);
    unsigned blk = idx >> 6;                                // NF4 block (64 elems)

    float scale = (float)q_scalers[blk] / q_factor[blk >> 8] + scaler_mean[0];
    float w = NF4_TAB[nf4_idx[idx] & 15] * scale;

    float acc = 0.0f;
#pragma unroll
    for (int r = 0; r < RANK; ++r)
        acc = fmaf(lora_b[o * RANK + r], lora_a[r * IN_DIM + i], acc);
    w = fmaf(LORA_SCALE, acc, w);

    weff[idx] = (__bf16)w;
}

// ---------------------------------------------------------------------------
// Kernel 2: x f32 -> bf16 (vectorized: float4 in, packed b64 out)
// ---------------------------------------------------------------------------
__global__ __launch_bounds__(256)
void cvt_bf16_kernel(const float4* __restrict__ x, __bf16* __restrict__ y)
{
    unsigned t = blockIdx.x * blockDim.x + threadIdx.x;
    float4 v = x[t];
    union { __bf16 h[4]; uint2 u; } pk;
    pk.h[0] = (__bf16)v.x; pk.h[1] = (__bf16)v.y;
    pk.h[2] = (__bf16)v.z; pk.h[3] = (__bf16)v.w;
    ((uint2*)y)[t] = pk.u;
}

// ---------------------------------------------------------------------------
// Kernel 3: GEMM  out[M, N] = X[M, K] @ W[N, K]^T   (bf16 in, f32 out)
// Block: 128x128, 256 threads = 8 waves in a 4(M) x 2(N) grid; each wave owns
// a 32x64 tile = 2x4 WMMA accumulators.  K stepped by 32.
//
// WMMA bf16 fragment layouts (wave32, per CDNA5 ISA 7.12.2):
//   A 16x32: lanes 0-15 -> M=lane, elems 0..7 = K 0..7,  8..15 = K 16..23
//            lanes16-31 -> M=lane-16, elems 0..7 = K 8..15, 8..15 = K 24..31
//   B 32x16: lanes 0-15 -> N=lane, elems 0..7 = K 0..7,  8..15 = K 16..23
//            lanes16-31 -> N=lane-16, elems 0..7 = K 8..15, 8..15 = K 24..31
// ---------------------------------------------------------------------------
__global__ __launch_bounds__(256)
void gemm_wmma_kernel(const __bf16* __restrict__ X,   // [M_TOTAL, IN_DIM]
                      const __bf16* __restrict__ W,   // [OUT_DIM, IN_DIM]
                      float*        __restrict__ out) // [M_TOTAL, OUT_DIM]
{
    __shared__ __align__(16) __bf16 ldsX[2 * XBUF];
    __shared__ __align__(16) __bf16 ldsW[2 * WBUF];

    const unsigned tid    = threadIdx.x;
    const unsigned lane   = tid & 31u;
    const unsigned wid    = tid >> 5;        // 0..7
    const unsigned half   = lane >> 4;       // 0 or 1
    const unsigned l16    = lane & 15u;
    const unsigned wave_m = wid >> 1;        // 0..3 -> 32-row slab
    const unsigned wave_n = wid & 1u;        // 0..1 -> 64-col slab

    const unsigned m_base = blockIdx.y * BM;
    const unsigned n_base = blockIdx.x * BN;

    // staging coordinates: each thread moves one b128 (8 bf16) per pass
    const unsigned srow = tid >> 2;          // 0..63
    const unsigned scol = (tid & 3u) * 8;    // 0,8,16,24

    // Issue async DMA fills of one 128x32 X tile and one 128x32 W tile into
    // buffer at element offset `bo`.  LDS operand of the async op = low 32
    // bits of the flat shared address (= LDS byte offset).
    auto issue_tile = [&](unsigned k0, unsigned boX, unsigned boW) {
#pragma unroll
        for (int p = 0; p < 2; ++p) {
            unsigned row = p * 64 + srow;
            unsigned           ldx = (unsigned)(uintptr_t)&ldsX[boX + row * BK + scol];
            unsigned long long gx  =
                (unsigned long long)(X + (size_t)(m_base + row) * IN_DIM + k0 + scol);
            asm volatile("global_load_async_to_lds_b128 %0, %1, off"
                         :: "v"(ldx), "v"(gx) : "memory");
            unsigned           ldw = (unsigned)(uintptr_t)&ldsW[boW + row * BK + scol];
            unsigned long long gw  =
                (unsigned long long)(W + (size_t)(n_base + row) * IN_DIM + k0 + scol);
            asm volatile("global_load_async_to_lds_b128 %0, %1, off"
                         :: "v"(ldw), "v"(gw) : "memory");
        }
    };

    const v8f vzero = {0.f,0.f,0.f,0.f,0.f,0.f,0.f,0.f};
    v8f acc[2][4];
#pragma unroll
    for (int mt = 0; mt < 2; ++mt)
#pragma unroll
        for (int nt = 0; nt < 4; ++nt)
            acc[mt][nt] = vzero;

    union Frag { v16bf v; uint4 q[2]; };

    // consume buffer at element offsets (boX, boW): 12 ds_load_b128 + 8 WMMA
    auto compute_tile = [&](unsigned boX, unsigned boW) {
        Frag fa[2], fb[4];
#pragma unroll
        for (int mt = 0; mt < 2; ++mt) {
            const uint4* rp =
                (const uint4*)&ldsX[boX + (wave_m * 32 + mt * 16 + l16) * BK];
            fa[mt].q[0] = rp[half];          // K = half*8 .. +7
            fa[mt].q[1] = rp[2 + half];      // K = 16 + half*8 .. +7
        }
#pragma unroll
        for (int nt = 0; nt < 4; ++nt) {
            const uint4* rp =
                (const uint4*)&ldsW[boW + (wave_n * 64 + nt * 16 + l16) * BK];
            fb[nt].q[0] = rp[half];          // K = half*8 .. +7
            fb[nt].q[1] = rp[2 + half];      // K = 16 + half*8 .. +7
        }
#pragma unroll
        for (int mt = 0; mt < 2; ++mt)
#pragma unroll
            for (int nt = 0; nt < 4; ++nt)
                acc[mt][nt] = __builtin_amdgcn_wmma_f32_16x16x32_bf16(
                    /*neg_a=*/false, fa[mt].v,
                    /*neg_b=*/false, fb[nt].v,
                    /*c_mod=*/(short)0, acc[mt][nt],
                    /*reuse_a=*/false, /*reuse_b=*/false);
    };

    issue_tile(0, 0, 0);                       // prologue: fill buffer 0

    unsigned boX = 0, boW = 0;
#pragma unroll 1
    for (int kt = 0; kt < KSTEPS - 1; ++kt) {
        // my async writes done, then all waves rendezvous (reads of the
        // buffer being refilled finished before each wave signaled)
        asm volatile("s_wait_asynccnt 0x0" ::: "memory");
        __syncthreads();
        issue_tile((unsigned)(kt + 1) * BK, boX ^ XBUF, boW ^ WBUF);
        compute_tile(boX, boW);
        boX ^= XBUF;
        boW ^= WBUF;
    }
    asm volatile("s_wait_asynccnt 0x0" ::: "memory");
    __syncthreads();
    compute_tile(boX, boW);                    // last tile, nothing to prefetch

    // Epilogue: C/D layout — lane n = l16, VGPR r -> M = r + half*8
#pragma unroll
    for (int mt = 0; mt < 2; ++mt) {
#pragma unroll
        for (int nt = 0; nt < 4; ++nt) {
            unsigned n  = n_base + wave_n * 64 + nt * 16 + l16;
            unsigned m0 = m_base + wave_m * 32 + mt * 16 + half * 8;
#pragma unroll
            for (int r = 0; r < 8; ++r)
                out[(size_t)(m0 + r) * OUT_DIM + n] = acc[mt][nt][r];
        }
    }
}

// ---------------------------------------------------------------------------
// Launch
// ---------------------------------------------------------------------------
extern "C" void kernel_launch(void* const* d_in, const int* in_sizes, int n_in,
                              void* d_out, int out_size, void* d_ws, size_t ws_size,
                              hipStream_t stream)
{
    const float* x           = (const float*)d_in[0];   // [4,2048,4096] f32
    const int*   nf4_idx     = (const int*)  d_in[1];   // [262144,64] i32
    const int*   q_scalers   = (const int*)  d_in[2];   // [262144] i32
    const float* q_factor    = (const float*)d_in[3];   // [1024] f32
    const float* scaler_mean = (const float*)d_in[4];   // [1] f32
    const float* lora_a      = (const float*)d_in[5];   // [16,4096] f32
    const float* lora_b      = (const float*)d_in[6];   // [4096,16] f32
    float*       out         = (float*)d_out;           // [8192,4096] f32

    // workspace: W_eff bf16 (32 MB) then x bf16 (64 MB)
    __bf16* weff = (__bf16*)d_ws;
    __bf16* xbf  = (__bf16*)((char*)d_ws + (size_t)OUT_DIM * IN_DIM * sizeof(__bf16));

    dequant_fold_kernel<<<(OUT_DIM * IN_DIM) / 256, 256, 0, stream>>>(
        nf4_idx, q_scalers, q_factor, scaler_mean, lora_a, lora_b, weff);

    cvt_bf16_kernel<<<((size_t)M_TOTAL * IN_DIM / 4) / 256, 256, 0, stream>>>(
        (const float4*)x, xbf);

    dim3 grid(OUT_DIM / BN, M_TOTAL / BM);   // (32, 64)
    gemm_wmma_kernel<<<grid, 256, 0, stream>>>(xbf, weff, out);
}
